// GPT2Attention_79809082294556
// MI455X (gfx1250) — matrix operand order repaired and checked by reading, not compiled, not run on hardware
//
#include <hip/hip_runtime.h>

// ---------------------------------------------------------------------------
// GPT-2 attention block for MI455X (gfx1250), bf16 WMMA pipeline.
// B=2, T=2048, C=1024, H=16, D=64.  ~103 GFLOP vs ~50 MB traffic at 23.3 TB/s
// -> compute-bound; all GEMM math routed through v_wmma_f32_16x16x32_bf16.
// GEMMs use a double-buffered LDS pipeline (1 barrier / K-step): next tile's
// global loads are issued before the barrier and only consumed (cvt+ds_store)
// after the WMMA phase, so load latency hides under compute.  Projection
// A-tiles stream via GLOBAL_LOAD_ASYNC_TO_LDS_B128 with ASYNCcnt pipelining.
// ---------------------------------------------------------------------------

typedef __bf16 bf16_t;
typedef __attribute__((ext_vector_type(16))) __bf16 v16bf;
typedef __attribute__((ext_vector_type(4)))  __bf16 v4bf;
typedef __attribute__((ext_vector_type(8)))  float  v8f;
typedef __attribute__((address_space(3))) void lds_void_t;

#define N_EMBD 1024
#define N_HEAD 16
#define HEADD  64
#define BB     2
#define TT     2048

__device__ __forceinline__ v8f wmma_bf16(v16bf a, v16bf b, v8f c) {
  return __builtin_amdgcn_wmma_f32_16x16x32_bf16(
      /*neg_a=*/false, a, /*neg_b=*/false, b,
      /*c_mod=*/(short)0, c, /*reuse_a=*/false, /*reuse_b=*/false);
}

// Async 16B global->LDS copy (gfx1250 GLOBAL_LOAD_ASYNC_TO_LDS_B128, ASYNCcnt).
__device__ __forceinline__ void async_b128_to_lds(void* lds_ptr, const void* gptr) {
  uint32_t loff = (uint32_t)(uintptr_t)(lds_void_t*)lds_ptr;  // raw LDS offset
  asm volatile("global_load_async_to_lds_b128 %0, %1, off"
               :: "v"(loff), "v"((unsigned long long)(uintptr_t)gptr)
               : "memory");
}

// A-fragment (16x32, MxK), ISA 7.12.2: lane m=L&15, half=L>>4.
__device__ __forceinline__ v16bf load_frag_a(const bf16_t* __restrict__ p,
                                             int stride, int lane) {
  int m = lane & 15, half = lane >> 4;
  const bf16_t* row = p + (size_t)m * stride;
  v16bf a;
#pragma unroll
  for (int e = 0; e < 16; ++e) a[e] = row[16 * half + e];
  return a;
}

// B-fragment (32x16, KxN) from a K-contiguous ("transposed") layout:
// p points at [n][k], row stride `stride`; lane n=L&15, half=L>>4.
__device__ __forceinline__ v16bf load_frag_bt(const bf16_t* __restrict__ p,
                                              int stride, int lane) {
  int n = lane & 15, half = lane >> 4;
  const bf16_t* row = p + (size_t)n * stride + 16 * half;
  v16bf b;
#pragma unroll
  for (int e = 0; e < 16; ++e) b[e] = row[e];
  return b;
}

// ---------------------------------------------------------------------------
// Kernel 1: qkv = x @ W_attn + b_attn   (M=4096, N=3072, K=1024)
// 256 threads (8 waves), 128x128 tile, K-step 32, double-buffered LDS.
// Epilogue scatters: q,k -> [B,H,T,D] bf16 ; v -> [B,H,D,T] bf16.
// ---------------------------------------------------------------------------
__global__ void __launch_bounds__(256)
qkv_kernel(const float* __restrict__ x, const float* __restrict__ W,
           const float* __restrict__ bias, bf16_t* __restrict__ qo,
           bf16_t* __restrict__ ko, bf16_t* __restrict__ vTo) {
  __shared__ bf16_t At[2][128 * 40];  // [m][k] tiles, stride 40 (bank pad)
  __shared__ bf16_t Bt[2][128 * 40];  // transposed [n][k] tiles

  const int tid = threadIdx.x;
  const int lane = tid & 31, wv = tid >> 5;
  const int wm = wv & 1, wn = wv >> 1;      // wave grid 2(M) x 4(N)
  const int m0 = blockIdx.x * 128;
  const int n0 = blockIdx.y * 128;

  float4 ra[4], rb[4];
  auto loadA = [&](int k0) {
#pragma unroll
    for (int i = 0; i < 4; ++i) {
      int slot = tid + i * 256, row = slot >> 3, c4 = (slot & 7) * 4;
      ra[i] = *(const float4*)(x + (size_t)(m0 + row) * N_EMBD + k0 + c4);
    }
  };
  auto loadB = [&](int k0) {
#pragma unroll
    for (int i = 0; i < 4; ++i) {
      int slot = tid + i * 256, kk = slot >> 5, n4 = (slot & 31) * 4;
      rb[i] = *(const float4*)(W + (size_t)(k0 + kk) * 3072 + n0 + n4);
    }
  };
  auto storeAB = [&](int buf) {
#pragma unroll
    for (int i = 0; i < 4; ++i) {   // A: packed cvt + 8B LDS stores
      int slot = tid + i * 256, row = slot >> 3, c4 = (slot & 7) * 4;
      v4bf t;
      t[0] = (bf16_t)ra[i].x; t[1] = (bf16_t)ra[i].y;
      t[2] = (bf16_t)ra[i].z; t[3] = (bf16_t)ra[i].w;
      *(v4bf*)(At[buf] + row * 40 + c4) = t;
    }
#pragma unroll
    for (int i = 0; i < 4; ++i) {   // B: transpose-in-store
      int slot = tid + i * 256, kk = slot >> 5, n4 = (slot & 31) * 4;
      Bt[buf][(n4 + 0) * 40 + kk] = (bf16_t)rb[i].x;
      Bt[buf][(n4 + 1) * 40 + kk] = (bf16_t)rb[i].y;
      Bt[buf][(n4 + 2) * 40 + kk] = (bf16_t)rb[i].z;
      Bt[buf][(n4 + 3) * 40 + kk] = (bf16_t)rb[i].w;
    }
  };

  v8f acc[4][2] = {};
  loadA(0); loadB(0); storeAB(0);
  int buf = 0;
  for (int k0 = 0; k0 < N_EMBD; k0 += 32) {
    const bool hasNext = (k0 + 32) < N_EMBD;
    if (hasNext) { loadA(k0 + 32); loadB(k0 + 32); }   // in flight over compute
    __syncthreads();                                   // cur tile visible
    v16bf bfr[2];
#pragma unroll
    for (int nt = 0; nt < 2; ++nt)
      bfr[nt] = load_frag_bt(Bt[buf] + (wn * 32 + nt * 16) * 40, 40, lane);
#pragma unroll
    for (int mt = 0; mt < 4; ++mt) {
      v16bf afr = load_frag_a(At[buf] + (wm * 64 + mt * 16) * 40, 40, lane);
      acc[mt][0] = wmma_bf16(afr, bfr[0], acc[mt][0]);
      acc[mt][1] = wmma_bf16(afr, bfr[1], acc[mt][1]);
    }
    if (hasNext) storeAB(buf ^ 1);   // s_wait_loadcnt lands after WMMA phase
    buf ^= 1;
  }

  // epilogue: bias + scatter into q/k/vT (bf16)
  const int lane_n = lane & 15, halfm = lane >> 4;
#pragma unroll
  for (int nt = 0; nt < 2; ++nt) {
    int gn = n0 + wn * 32 + nt * 16 + lane_n;     // column in [0,3C)
    float bb = bias[gn];
    int which = gn >> 10;                         // 0=q 1=k 2=v
    int cc = gn & 1023;
    int h = cc >> 6, d = cc & 63;
#pragma unroll
    for (int mt = 0; mt < 4; ++mt) {
#pragma unroll
      for (int r = 0; r < 8; ++r) {
        int gm = m0 + wm * 64 + mt * 16 + r + 8 * halfm;  // row in [0,B*T)
        int bidx = gm >> 11;
        int t = gm & 2047;
        bf16_t bv = (bf16_t)(acc[mt][nt][r] + bb);
        size_t bh = (size_t)(bidx * N_HEAD + h);
        if (which == 0)      qo[(bh * TT + t) * HEADD + d] = bv;
        else if (which == 1) ko[(bh * TT + t) * HEADD + d] = bv;
        else                 vTo[(bh * HEADD + d) * TT + t] = bv;
      }
    }
  }
}

// ---------------------------------------------------------------------------
// Kernel 2: causal flash attention, bf16 WMMA, f32 accumulate.
// 128 threads (4 waves); each wave owns 16 query rows, block owns 64.
// 64-key blocks: 8 wmma (QK^T) + online softmax + 8 wmma (PV).
// ---------------------------------------------------------------------------
__global__ void __launch_bounds__(128)
attn_kernel(const bf16_t* __restrict__ q, const bf16_t* __restrict__ k,
            const bf16_t* __restrict__ vT, bf16_t* __restrict__ y) {
  __shared__ bf16_t plds[4][16 * 72];  // per-wave P tile, 16x64 (stride 72)

  const int lane = threadIdx.x & 31;
  const int wv = threadIdx.x >> 5;
  const int bh = blockIdx.x;               // b*H + h
  const int b = bh >> 4, h = bh & 15;
  const int qbase = blockIdx.y * 64 + wv * 16;
  const size_t bhT = (size_t)bh * TT;
  const size_t bhD = (size_t)bh * HEADD;
  const int half = lane >> 4, n = lane & 15;

  v16bf qa[2];
  qa[0] = load_frag_a(q + (bhT + qbase) * HEADD + 0, HEADD, lane);
  qa[1] = load_frag_a(q + (bhT + qbase) * HEADD + 32, HEADD, lane);

  v8f acc[4] = {};
  float mrow[8], lrow[8];
#pragma unroll
  for (int r = 0; r < 8; ++r) { mrow[r] = -3.0e30f; lrow[r] = 0.f; }

  const int nblocks = (qbase + 16 + 63) >> 6;  // causal bound (wave-uniform)
  for (int j = 0; j < nblocks; ++j) {
    const int kbase = j * 64;
    v8f s[4] = {};
#pragma unroll
    for (int kk = 0; kk < 2; ++kk) {
#pragma unroll
      for (int tc = 0; tc < 4; ++tc) {
        v16bf kb = load_frag_bt(k + (bhT + kbase + 16 * tc) * HEADD + 32 * kk,
                                HEADD, lane);
        s[tc] = wmma_bf16(qa[kk], kb, s[tc]);
      }
    }
    // causal mask + online softmax; P written straight to LDS (A-layout src)
    bf16_t* pl = &plds[wv][0];
#pragma unroll
    for (int r = 0; r < 8; ++r) {
      int tq = qbase + r + 8 * half;
      float vv[4];
#pragma unroll
      for (int tc = 0; tc < 4; ++tc) {
        float sv = s[tc][r] * 0.125f;
        vv[tc] = (kbase + 16 * tc + n > tq) ? -3.0e30f : sv;
      }
      float mx = fmaxf(fmaxf(vv[0], vv[1]), fmaxf(vv[2], vv[3]));
#pragma unroll
      for (int d = 1; d < 16; d <<= 1) mx = fmaxf(mx, __shfl_xor(mx, d, 32));
      float mnew = fmaxf(mrow[r], mx);
      float alpha = __expf(mrow[r] - mnew);
      mrow[r] = mnew;
      float sum = 0.f;
#pragma unroll
      for (int tc = 0; tc < 4; ++tc) {
        float e = __expf(vv[tc] - mnew);
        sum += e;
        pl[(r + 8 * half) * 72 + 16 * tc + n] = (bf16_t)e;
      }
#pragma unroll
      for (int d = 1; d < 16; d <<= 1) sum += __shfl_xor(sum, d, 32);
      lrow[r] = lrow[r] * alpha + sum;
#pragma unroll
      for (int dt = 0; dt < 4; ++dt) acc[dt][r] *= alpha;
    }
    // reshape P (C-layout -> A-layout) via wave-private LDS (in-order per wave)
    asm volatile("s_wait_dscnt 0" ::: "memory");
    v16bf pa0 = load_frag_a(pl, 72, lane);        // keys kbase..kbase+31
    v16bf pa1 = load_frag_a(pl + 32, 72, lane);   // keys kbase+32..kbase+63
    asm volatile("" ::: "memory");
#pragma unroll
    for (int dt = 0; dt < 4; ++dt) {
      const bf16_t* vrow = vT + (bhD + dt * 16) * TT + kbase;
      v16bf vb0 = load_frag_bt(vrow, TT, lane);
      v16bf vb1 = load_frag_bt(vrow + 32, TT, lane);
      acc[dt] = wmma_bf16(pa0, vb0, acc[dt]);
      acc[dt] = wmma_bf16(pa1, vb1, acc[dt]);
    }
  }

  // epilogue: y = acc / l  (one rcp per row; bf16 for the projection GEMM)
#pragma unroll
  for (int r = 0; r < 8; ++r) {
    int tq = qbase + r + 8 * half;
    float inv = 1.0f / lrow[r];
#pragma unroll
    for (int dt = 0; dt < 4; ++dt) {
      y[((size_t)b * TT + tq) * N_EMBD + h * HEADD + dt * 16 + n] =
          (bf16_t)(acc[dt][r] * inv);
    }
  }
}

// ---------------------------------------------------------------------------
// Kernel 3: out = y @ W_proj + b_proj   (M=4096, N=1024, K=1024), f32 out.
// A-tile (bf16) streams via async DMA to LDS, pipelined on ASYNCcnt;
// B-tile double-buffered through registers like kernel 1.
// ---------------------------------------------------------------------------
__global__ void __launch_bounds__(256)
proj_kernel(const bf16_t* __restrict__ y, const float* __restrict__ W,
            const float* __restrict__ bias, float* __restrict__ out) {
  __shared__ bf16_t At[2][128 * 40];
  __shared__ bf16_t Bt[2][128 * 40];

  const int tid = threadIdx.x;
  const int lane = tid & 31, wv = tid >> 5;
  const int wm = wv & 1, wn = wv >> 1;
  const int m0 = blockIdx.x * 128;
  const int n0 = blockIdx.y * 128;

  float4 rb[4];
  auto stageA_async = [&](int buf, int k0) {   // 2 x 16B DMA per thread
#pragma unroll
    for (int i = 0; i < 2; ++i) {
      int slot = tid + i * 256, row = slot >> 2, c8 = (slot & 3) * 8;
      async_b128_to_lds(At[buf] + row * 40 + c8,
                        y + (size_t)(m0 + row) * N_EMBD + k0 + c8);
    }
  };
  auto loadB = [&](int k0) {
#pragma unroll
    for (int i = 0; i < 4; ++i) {
      int slot = tid + i * 256, kk = slot >> 5, n4 = (slot & 31) * 4;
      rb[i] = *(const float4*)(W + (size_t)(k0 + kk) * N_EMBD + n0 + n4);
    }
  };
  auto storeB = [&](int buf) {
#pragma unroll
    for (int i = 0; i < 4; ++i) {
      int slot = tid + i * 256, kk = slot >> 5, n4 = (slot & 31) * 4;
      Bt[buf][(n4 + 0) * 40 + kk] = (bf16_t)rb[i].x;
      Bt[buf][(n4 + 1) * 40 + kk] = (bf16_t)rb[i].y;
      Bt[buf][(n4 + 2) * 40 + kk] = (bf16_t)rb[i].z;
      Bt[buf][(n4 + 3) * 40 + kk] = (bf16_t)rb[i].w;
    }
  };

  v8f acc[4][2] = {};
  stageA_async(0, 0);
  loadB(0); storeB(0);
  int buf = 0;
  for (int k0 = 0; k0 < N_EMBD; k0 += 32) {
    const bool hasNext = (k0 + 32) < N_EMBD;
    if (hasNext) {
      stageA_async(buf ^ 1, k0 + 32);            // next A DMA in flight
      loadB(k0 + 32);                            // next B loads in flight
      asm volatile("s_wait_asynccnt 0x2" ::: "memory");  // cur A done, next pending
    } else {
      asm volatile("s_wait_asynccnt 0x0" ::: "memory");  // drain last A
    }
    __syncthreads();
    v16bf bfr[2];
#pragma unroll
    for (int nt = 0; nt < 2; ++nt)
      bfr[nt] = load_frag_bt(Bt[buf] + (wn * 32 + nt * 16) * 40, 40, lane);
#pragma unroll
    for (int mt = 0; mt < 4; ++mt) {
      v16bf afr = load_frag_a(At[buf] + (wm * 64 + mt * 16) * 40, 40, lane);
      acc[mt][0] = wmma_bf16(afr, bfr[0], acc[mt][0]);
      acc[mt][1] = wmma_bf16(afr, bfr[1], acc[mt][1]);
    }
    if (hasNext) storeB(buf ^ 1);
    buf ^= 1;
  }

  const int lane_n = lane & 15, halfm = lane >> 4;
#pragma unroll
  for (int nt = 0; nt < 2; ++nt) {
    int gn = n0 + wn * 32 + nt * 16 + lane_n;
    float bb = bias[gn];
#pragma unroll
    for (int mt = 0; mt < 4; ++mt) {
#pragma unroll
      for (int r = 0; r < 8; ++r) {
        int gm = m0 + wm * 64 + mt * 16 + r + 8 * halfm;
        out[(size_t)gm * N_EMBD + gn] = acc[mt][nt][r] + bb;
      }
    }
  }
}

// ---------------------------------------------------------------------------
// Launch: x,W_attn,b_attn,W_proj,b_proj (all f32) -> out f32 [B,T,C].
// Workspace (32 MB): q(8MB) k(8MB) vT(8MB) bf16 + y(8MB) bf16.
// ---------------------------------------------------------------------------
extern "C" void kernel_launch(void* const* d_in, const int* in_sizes, int n_in,
                              void* d_out, int out_size, void* d_ws, size_t ws_size,
                              hipStream_t stream) {
  const float* x      = (const float*)d_in[0];
  const float* W_attn = (const float*)d_in[1];
  const float* b_attn = (const float*)d_in[2];
  const float* W_proj = (const float*)d_in[3];
  const float* b_proj = (const float*)d_in[4];
  float* out = (float*)d_out;

  char* ws = (char*)d_ws;
  const size_t MB = 1024ull * 1024ull;
  bf16_t* q  = (bf16_t*)(ws + 0 * MB);
  bf16_t* k  = (bf16_t*)(ws + 8 * MB);
  bf16_t* vT = (bf16_t*)(ws + 16 * MB);
  bf16_t* yb = (bf16_t*)(ws + 24 * MB);

  qkv_kernel<<<dim3(32, 24), 256, 0, stream>>>(x, W_attn, b_attn, q, k, vT);
  attn_kernel<<<dim3(32, 32), 128, 0, stream>>>(q, k, vT, yb);
  proj_kernel<<<dim3(32, 8), 256, 0, stream>>>(yb, W_proj, b_proj, out);
}
